// RadarTransformerNet_26431228739631
// MI455X (gfx1250) — matrix operand order repaired and checked by reading, not compiled, not run on hardware
//
#include <hip/hip_runtime.h>
#include <hip/hip_bf16.h>
#include <math.h>

// ======================= CDNA5 WMMA plumbing =======================
typedef __attribute__((ext_vector_type(16))) __bf16 v16bf;
typedef __attribute__((ext_vector_type(8)))  float  v8f;

union Frag16 { unsigned int u[8]; v16bf v; };

__device__ __forceinline__ unsigned short f2bf(float f) {
  unsigned int u = __float_as_uint(f);
  u += 0x7FFFu + ((u >> 16) & 1u);          // round-to-nearest-even
  return (unsigned short)(u >> 16);
}
__device__ __forceinline__ float bf2f(unsigned short h) {
  return __uint_as_float(((unsigned int)h) << 16);
}
__device__ __forceinline__ unsigned int pack2bf(float a, float b) {
  return (unsigned int)f2bf(a) | ((unsigned int)f2bf(b) << 16);
}
// A-fragment (16x32 bf16) K of low element in 32-bit slot j, per ISA 7.12.2 table
__device__ __forceinline__ int kslotA(int half, int j) {
  return half ? ((j < 4) ? (8 + 2*j) : (24 + 2*(j-4)))
              : ((j < 4) ? (2*j)     : (16 + 2*(j-4)));
}
// B-fragment (32x16 bf16): same K order as A-fragment (ISA 7.12.2 table)
__device__ __forceinline__ int kslotB(int half, int j) { return kslotA(half, j); }

__device__ __forceinline__ v8f wmma_bf16(const Frag16& a, const Frag16& b, v8f c) {
  return __builtin_amdgcn_wmma_f32_16x16x32_bf16(false, a.v, false, b.v,
                                                 (short)0, c, false, false);
}

// CDNA5 async DMA: global -> LDS, per-lane addresses, tracked by ASYNCcnt
__device__ __forceinline__ void async_g2l_b128(unsigned lds_addr, const float* gaddr) {
  asm volatile("global_load_async_to_lds_b128 %0, %1, off"
               :: "v"(lds_addr), "v"(gaddr) : "memory");
}
__device__ __forceinline__ void wait_asynccnt0() {
  asm volatile("s_wait_asynccnt 0x0" ::: "memory");
}

// ======================= elementwise / stem ========================
__global__ void transpose_in(const float* __restrict__ x, float* __restrict__ y) {
  // (B,4,128,256,2) -> (B,2,4,128,256)
  const long total = 2L*2*4*128*256;
  for (long idx = (long)blockIdx.x*256 + threadIdx.x; idx < total;
       idx += (long)gridDim.x*256) {
    long t = idx;
    int sa = t % 256; t /= 256;
    int ch = t % 128; t /= 128;
    int rx = t % 4;   t /= 4;
    int c2 = t % 2;   int b = (int)(t / 2);
    y[idx] = x[((((size_t)b*4 + rx)*128 + ch)*256 + sa)*2 + c2];
  }
}

__global__ void conv3d_bn_relu(const float* __restrict__ in, const float* __restrict__ wt,
    const float* __restrict__ cbias, const float* __restrict__ gg, const float* __restrict__ be,
    float* __restrict__ out, int Bn, int Ci, int Di, int Hi, int Wi,
    int Co, int Do, int Ho, int Wo, int kd, int kh, int kw,
    int sd, int sh, int sw, int pd, int ph, int pw) {
  const long total = (long)Bn*Co*Do*Ho*Wo;
  const float bnscale = rsqrtf(1.0f + 1e-5f);   // reference _bn uses 1/sqrt(1+eps)
  for (long idx = (long)blockIdx.x*256 + threadIdx.x; idx < total;
       idx += (long)gridDim.x*256) {
    long t = idx;
    int ow = t % Wo; t /= Wo;
    int oh = t % Ho; t /= Ho;
    int od = t % Do; t /= Do;
    int co = t % Co; int b = (int)(t / Co);
    float acc = 0.f;
    const float* wte = wt + (size_t)co*Ci*kd*kh*kw;
    for (int ci = 0; ci < Ci; ++ci) {
      const float* ip = in + ((size_t)b*Ci + ci)*Di*Hi*Wi;
      for (int z = 0; z < kd; ++z) {
        int id = od*sd + z - pd; if ((unsigned)id >= (unsigned)Di) continue;
        for (int y = 0; y < kh; ++y) {
          int ih = oh*sh + y - ph; if ((unsigned)ih >= (unsigned)Hi) continue;
          const float* row  = ip + ((size_t)id*Hi + ih)*Wi;
          const float* wrow = wte + ((size_t)(ci*kd + z)*kh + y)*kw;
          for (int xk = 0; xk < kw; ++xk) {
            int iw = ow*sw + xk - pw; if ((unsigned)iw >= (unsigned)Wi) continue;
            acc += row[iw] * wrow[xk];
          }
        }
      }
    }
    float v = (acc + cbias[co]) * (gg[co]*bnscale) + be[co];
    out[idx] = fmaxf(v, 0.f);
  }
}

__global__ void add_posemb(const float* __restrict__ h, const float* __restrict__ pos,
                           float* __restrict__ y, int C, int HW, long total) {
  for (long idx = (long)blockIdx.x*256 + threadIdx.x; idx < total;
       idx += (long)gridDim.x*256)
    y[idx] = h[idx] + pos[(idx / HW) % C];
}
__global__ void add_ip(float* __restrict__ x, const float* __restrict__ a, long n) {
  for (long i = (long)blockIdx.x*256 + threadIdx.x; i < n; i += (long)gridDim.x*256)
    x[i] += a[i];
}
__global__ void add2_ip(float* __restrict__ x, const float* __restrict__ a,
                        const float* __restrict__ b, long n) {
  for (long i = (long)blockIdx.x*256 + threadIdx.x; i < n; i += (long)gridDim.x*256)
    x[i] += a[i] + b[i];
}

// ============ per-batch LayerNorm (reference _gn over axes 1,2,3) ============
__global__ void __launch_bounds__(256) gn_ln(const float* __restrict__ X,
    const float* __restrict__ g, const float* __restrict__ bb, float* __restrict__ Y,
    int C, int HW) {
  const int b = blockIdx.x;
  const int n = C*HW;
  const float* xb = X + (size_t)b*n;
  float* yb = Y + (size_t)b*n;
  __shared__ float red[256], red2[256];
  float s = 0.f, s2 = 0.f;
  for (int i = threadIdx.x; i < n; i += 256) { float v = xb[i]; s += v; s2 += v*v; }
  red[threadIdx.x] = s; red2[threadIdx.x] = s2;
  __syncthreads();
  for (int off = 128; off > 0; off >>= 1) {
    if ((int)threadIdx.x < off) { red[threadIdx.x] += red[threadIdx.x+off];
                                  red2[threadIdx.x] += red2[threadIdx.x+off]; }
    __syncthreads();
  }
  float mu  = red[0] / (float)n;
  float var = red2[0] / (float)n - mu*mu;
  float inv = rsqrtf(var + 1e-5f);
  for (int i = threadIdx.x; i < n; i += 256) {
    int c = i / HW;
    yb[i] = (xb[i] - mu) * inv * g[c] + bb[c];
  }
}

// ================== WMMA pointwise GEMM (1x1 conv / MLP) ==================
// Y[b,o,hw] = act( sum_c Wt[o,c]*X[b,c,hw] + bias[o] );  act: 0=none, 1=exact GELU
// WG tile: 64 tokens x 64 outputs; wave tile: 16 tokens x 32 outputs (2 accs, shared A-frag)
__global__ void __launch_bounds__(256) gemm_pw(const float* __restrict__ X,
    const float* __restrict__ Wt, const float* __restrict__ bias,
    float* __restrict__ Y, int C, int O, int HW, int act) {
  const int lane = threadIdx.x, wave = threadIdx.y;
  const int tid  = wave*32 + lane;
  const int half = lane >> 4, ln = lane & 15;
  const int t0 = blockIdx.x * 64;
  const int o0 = blockIdx.y * 64;
  const int b  = blockIdx.z;
  const float* Xb = X + (size_t)b * C * HW;
  float* Yb = Y + (size_t)b * O * HW;
  const int wm = wave & 3, wn = wave >> 2;

  __shared__ unsigned short Xs[64][32];   // [token][k] bf16
  __shared__ unsigned short Ws[64][32];   // [out][k]  bf16

  v8f acc0 = {0.f,0.f,0.f,0.f,0.f,0.f,0.f,0.f};
  v8f acc1 = {0.f,0.f,0.f,0.f,0.f,0.f,0.f,0.f};
  for (int c0 = 0; c0 < C; c0 += 32) {
    __syncthreads();
    {
      int idx = tid;
      for (int i = 0; i < 8; ++i, idx += 256) {       // 64x32 X tile
        int tt = idx & 63, kk = idx >> 6;
        Xs[tt][kk] = f2bf(Xb[(size_t)(c0+kk)*HW + t0 + tt]);
      }
      idx = tid;
      for (int i = 0; i < 8; ++i, idx += 256) {       // 64x32 W tile
        int kk = idx & 31, oo = idx >> 5;
        Ws[oo][kk] = f2bf(Wt[(size_t)(o0+oo)*C + c0 + kk]);
      }
      if (c0 + 32 < C) {                              // hint next K-tile into caches
        __builtin_prefetch(Xb + (size_t)(c0 + 32 + (tid >> 6))*HW + t0 + (tid & 63), 0, 3);
        __builtin_prefetch(Wt + (size_t)(o0 + (tid >> 2))*C + c0 + 32, 0, 3);
      }
    }
    __syncthreads();
    Frag16 fa, fb0, fb1;
    const unsigned short* ra  = &Xs[wm*16 + ln][0];
    const unsigned short* rb0 = &Ws[wn*32 + ln][0];
    const unsigned short* rb1 = &Ws[wn*32 + 16 + ln][0];
    for (int j = 0; j < 8; ++j) {
      fa.u[j]  = *(const unsigned int*)(ra  + kslotA(half, j));
      fb0.u[j] = *(const unsigned int*)(rb0 + kslotB(half, j));
      fb1.u[j] = *(const unsigned int*)(rb1 + kslotB(half, j));
    }
    acc0 = wmma_bf16(fa, fb0, acc0);
    acc1 = wmma_bf16(fa, fb1, acc1);
  }
  for (int r = 0; r < 8; ++r) {
    int tok = t0 + wm*16 + r + 8*half;
    int oA  = o0 + wn*32 + ln;
    int oB  = oA + 16;
    float vA = acc0[r] + bias[oA];
    float vB = acc1[r] + bias[oB];
    if (act == 1) {
      vA = 0.5f * vA * (1.0f + erff(vA * 0.70710678118654752f));
      vB = 0.5f * vB * (1.0f + erff(vB * 0.70710678118654752f));
    }
    Yb[(size_t)oA*HW + tok] = vA;
    Yb[(size_t)oB*HW + tok] = vB;
  }
}

// ===================== flash attention (full MHA) =====================
// heads=8, d=32, N=HW tokens. qkv layout (B, 3*256, HW); out (B, 256, HW).
// K/V tiles staged with CDNA5 async global->LDS DMA (ASYNCcnt).
#define FA_QT 128
#define FA_KT 64
__global__ void __launch_bounds__(256) flash_attn(const float* __restrict__ qkv,
                                                  float* __restrict__ out, int HW) {
  const int lane = threadIdx.x, wave = threadIdx.y;
  const int tid  = wave*32 + lane;
  const int half = lane >> 4, ln = lane & 15;
  const int qt0 = blockIdx.x * FA_QT;
  const int hed = blockIdx.y;
  const int b   = blockIdx.z;
  const float scale = 0.17677669529663687f;     // 32^-0.5
  const float* qp = qkv + ((size_t)b*768 + hed*32) * HW;
  const float* kp = qp + (size_t)256*HW;
  const float* vp = qp + (size_t)512*HW;

  __shared__ unsigned short Qs[FA_QT][32];              // [q][d] bf16   8 KB
  __shared__ __align__(16) float Ksf[32][FA_KT];        // [d][key] f32  8 KB
  __shared__ __align__(16) float Vtf[32][FA_KT];        // [d][key] f32  8 KB
  __shared__ unsigned short Ps[8][16][FA_KT];           // per-wave P   16 KB

  for (int i = 0; i < 16; ++i) {                // stage Q (pre-scaled, bf16)
    int idx = tid + i*256;
    int q = idx & 127, dd = idx >> 7;
    Qs[q][dd] = f2bf(qp[(size_t)dd*HW + qt0 + q] * scale);
  }
  __syncthreads();

  Frag16 aQ;                                    // Q fragment is loop-invariant
  {
    const unsigned short* rp = &Qs[wave*16 + ln][0];
    for (int j = 0; j < 8; ++j) aQ.u[j] = *(const unsigned int*)(rp + kslotA(half, j));
  }

  float rmax[8], rsum[8];
  v8f o0 = {0.f,0.f,0.f,0.f,0.f,0.f,0.f,0.f};
  v8f o1 = {0.f,0.f,0.f,0.f,0.f,0.f,0.f,0.f};
  for (int r = 0; r < 8; ++r) { rmax[r] = -1e30f; rsum[r] = 0.f; }

  const int ntiles = HW / FA_KT;
  for (int kt = 0; kt < ntiles; ++kt) {
    const int kb = kt * FA_KT;
    __syncthreads();
    {                                            // async DMA K,V -> LDS (f32)
      int idx = tid;
      for (int i = 0; i < 2; ++i, idx += 256) {  // 512 x b128 each for K and V
        int dd = idx >> 4, ky = (idx & 15) * 4;
        async_g2l_b128((unsigned)(size_t)&Ksf[dd][ky], kp + (size_t)dd*HW + kb + ky);
        async_g2l_b128((unsigned)(size_t)&Vtf[dd][ky], vp + (size_t)dd*HW + kb + ky);
      }
      wait_asynccnt0();
    }
    __syncthreads();

    v8f s[4];                                    // S = Q K^T  (K-dim = d = 32)
    for (int nt = 0; nt < 4; ++nt) {
      Frag16 bK;
      const int key = nt*16 + ln;
      for (int j = 0; j < 8; ++j) {
        int d0 = kslotB(half, j);
        bK.u[j] = pack2bf(Ksf[d0][key], Ksf[d0+1][key]);
      }
      v8f z = {0.f,0.f,0.f,0.f,0.f,0.f,0.f,0.f};
      s[nt] = wmma_bf16(aQ, bK, z);
    }

    // online softmax; C-frag row r maps to query row (r + 8*half)
    for (int r = 0; r < 8; ++r) {
      float tm = fmaxf(fmaxf(s[0][r], s[1][r]), fmaxf(s[2][r], s[3][r]));
      for (int off = 1; off < 16; off <<= 1) tm = fmaxf(tm, __shfl_xor(tm, off, 32));
      float mnew  = fmaxf(rmax[r], tm);
      float alpha = __expf(rmax[r] - mnew);
      rmax[r] = mnew;
      float ps = 0.f;
      for (int nt = 0; nt < 4; ++nt) {
        float p = __expf(s[nt][r] - mnew);
        s[nt][r] = p; ps += p;
      }
      for (int off = 1; off < 16; off <<= 1) ps += __shfl_xor(ps, off, 32);
      rsum[r] = rsum[r]*alpha + ps;
      o0[r] *= alpha; o1[r] *= alpha;
    }

    // spill P (bf16) to per-wave LDS to re-fragment C-layout -> A-layout
    for (int nt = 0; nt < 4; ++nt)
      for (int r = 0; r < 8; ++r)
        Ps[wave][r + 8*half][nt*16 + ln] = f2bf(s[nt][r]);

    for (int kk = 0; kk < 2; ++kk) {            // O += P V  (K = 64 keys)
      Frag16 aP, bV0, bV1;
      const unsigned short* rp = &Ps[wave][ln][kk*32];
      for (int j = 0; j < 8; ++j)
        aP.u[j] = *(const unsigned int*)(rp + kslotA(half, j));
      for (int j = 0; j < 8; ++j) {
        int k0 = kk*32 + kslotB(half, j);
        const float2* p0 = (const float2*)&Vtf[ln][k0];        // d = ln
        const float2* p1 = (const float2*)&Vtf[16 + ln][k0];   // d = 16+ln
        bV0.u[j] = pack2bf(p0->x, p0->y);
        bV1.u[j] = pack2bf(p1->x, p1->y);
      }
      o0 = wmma_bf16(aP, bV0, o0);
      o1 = wmma_bf16(aP, bV1, o1);
    }
  }

  float* op = out + ((size_t)b*256 + hed*32) * HW;
  for (int r = 0; r < 8; ++r) {
    int tok = qt0 + wave*16 + r + 8*half;
    float inv = 1.f / rsum[r];
    op[(size_t)ln*HW + tok]        = o0[r] * inv;     // d = ln
    op[(size_t)(16 + ln)*HW + tok] = o1[r] * inv;     // d = 16+ln
  }
}

// ===================== axial attention (heads=4, d=64) =====================
// mode 0 = 'range'  (seq over W=32, rows=H=64); mode 1 = 'doppler' (seq over H=64, rows=W=32)
__global__ void __launch_bounds__(256) axial_attn(const float* __restrict__ qkv,
    float* __restrict__ out, int HW, int mode) {
  const int S = mode ? 64 : 32;
  const int R = mode ? 32 : 64;
  int wg = blockIdx.x;
  int r = wg % R; wg /= R;
  int hed = wg % 4; int b = wg / 4;
  const float* base = qkv + (size_t)b*768*HW;
  const int tid = threadIdx.x;

  __shared__ unsigned short qs[64*64], ks[64*64], vs[64*64];  // [dd][s], stride 64
  __shared__ float a[64*64];                                  // scores [i][j]

  for (int idx = tid; idx < 64*S; idx += 256) {
    int dd = idx / S, s = idx % S;
    int sp = mode ? (s*32 + r) : (r*32 + s);
    int ch = hed*64 + dd;
    qs[dd*64 + s] = f2bf(base[(size_t)ch*HW + sp]);
    ks[dd*64 + s] = f2bf(base[(size_t)(256 + ch)*HW + sp]);
    vs[dd*64 + s] = f2bf(base[(size_t)(512 + ch)*HW + sp]);
  }
  __syncthreads();

  for (int idx = tid; idx < S*S; idx += 256) {
    int i = idx / S, j = idx % S;
    float acc = 0.f;
    for (int dd = 0; dd < 64; ++dd)
      acc += bf2f(qs[dd*64 + i]) * bf2f(ks[dd*64 + j]);
    a[i*64 + j] = acc * 0.125f;                 // 64^-0.5
  }
  __syncthreads();

  if (tid < S) {                                 // softmax over row tid
    float mx = -1e30f;
    for (int j = 0; j < S; ++j) mx = fmaxf(mx, a[tid*64 + j]);
    float sm = 0.f;
    for (int j = 0; j < S; ++j) { float e = __expf(a[tid*64 + j] - mx); a[tid*64 + j] = e; sm += e; }
    float inv = 1.f / sm;
    for (int j = 0; j < S; ++j) a[tid*64 + j] *= inv;
  }
  __syncthreads();

  for (int idx = tid; idx < 64*S; idx += 256) {
    int dd = idx / S, ii = idx % S;
    float acc = 0.f;
    for (int j = 0; j < S; ++j) acc += a[ii*64 + j] * bf2f(vs[dd*64 + j]);
    size_t off;
    if (mode == 0)  // (B,d,heads,W,H) -> flat: c=dd*4+hed, sp = w*64 + h
      off = (size_t)b*256*HW + (size_t)(dd*4 + hed)*HW + (size_t)ii*64 + r;
    else            // (B,d,H,heads,W) -> flat
      off = (size_t)b*256*HW + ((size_t)(dd*64 + ii)*4 + hed)*32 + r;
    out[off] = acc;
  }
}

// ========================= DFT magnitude =========================
__global__ void __launch_bounds__(256) fft_mag(const float* __restrict__ h,
    const float* __restrict__ wr, const float* __restrict__ wd, float* __restrict__ out) {
  const int bc = blockIdx.x;                    // one (b,c) slab of [64 ch][32 sa]
  const float* xb = h + (size_t)bc*2048;
  __shared__ float xs[64*32];
  __shared__ float rr[64*32], ri[64*32];
  for (int i = threadIdx.x; i < 2048; i += 256) xs[i] = xb[i];
  __syncthreads();
  for (int i = threadIdx.x; i < 2048; i += 256) {
    int c = i >> 5, j = i & 31;
    float ar = 0.f, ai = 0.f;
    for (int s = 0; s < 32; ++s) {
      float xv = xs[c*32 + s];
      ar += xv * wr[(s*32 + j)*2 + 0];
      ai += xv * wr[(s*32 + j)*2 + 1];
    }
    rr[i] = ar; ri[i] = ai;
  }
  __syncthreads();
  for (int i = threadIdx.x; i < 2048; i += 256) {
    int ii = i >> 5, j = i & 31;
    float dr = 0.f, di = 0.f;
    for (int c = 0; c < 64; ++c) {
      float w0 = wd[(c*64 + ii)*2 + 0], w1 = wd[(c*64 + ii)*2 + 1];
      float r0 = rr[c*32 + j], i0 = ri[c*32 + j];
      dr += r0*w0 - i0*w1;
      di += r0*w1 + i0*w0;
    }
    out[(size_t)bc*2048 + i] = sqrtf(dr*dr + di*di) * (1.0f/2048.0f);
  }
}

// ===================== output head 1x1 + sigmoid =====================
__global__ void op2_sigmoid(const float* __restrict__ in, const float* __restrict__ w,
                            const float* __restrict__ bb, float* __restrict__ out) {
  int idx = blockIdx.x*256 + threadIdx.x;       // 2*64*32 = 4096
  if (idx >= 4096) return;
  int hw = idx % 2048, b = idx / 2048;
  const float* ib = in + (size_t)b*128*2048;
  float acc = bb[0];
  for (int c = 0; c < 128; ++c) acc += ib[(size_t)c*2048 + hw] * w[c];
  out[idx] = 1.0f / (1.0f + __expf(-acc));
}

// ============================ driver ============================
extern "C" void kernel_launch(void* const* d_in, const int* in_sizes, int n_in,
                              void* d_out, int out_size, void* d_ws, size_t ws_size,
                              hipStream_t stream) {
  (void)in_sizes; (void)n_in; (void)out_size; (void)ws_size;
  const float* x      = (const float*)d_in[0];
  const float* tp_w1  = (const float*)d_in[1];  const float* tp_b1 = (const float*)d_in[2];
  const float* tp_g1  = (const float*)d_in[3];  const float* tp_be1= (const float*)d_in[4];
  const float* tp_w2  = (const float*)d_in[5];  const float* tp_b2 = (const float*)d_in[6];
  const float* tp_g2  = (const float*)d_in[7];  const float* tp_be2= (const float*)d_in[8];
  const float* tp_w3  = (const float*)d_in[9];  const float* tp_b3 = (const float*)d_in[10];
  const float* tp_g3  = (const float*)d_in[11]; const float* tp_be3= (const float*)d_in[12];
  const float* cb_w1  = (const float*)d_in[13]; const float* cb_b1 = (const float*)d_in[14];
  const float* cb_g1  = (const float*)d_in[15]; const float* cb_be1= (const float*)d_in[16];
  const float* cb_w2  = (const float*)d_in[17]; const float* cb_b2 = (const float*)d_in[18];
  const float* cb_g2  = (const float*)d_in[19]; const float* cb_be2= (const float*)d_in[20];
  const float* cb_w3  = (const float*)d_in[21]; const float* cb_b3 = (const float*)d_in[22];
  const float* cb_g3  = (const float*)d_in[23]; const float* cb_be3= (const float*)d_in[24];
  const float* fft_wr = (const float*)d_in[25];
  const float* fft_wd = (const float*)d_in[26];
  const float* pos    = (const float*)d_in[27];
  const float* n1g = (const float*)d_in[28]; const float* n1b = (const float*)d_in[29];
  const float* n2g = (const float*)d_in[30]; const float* n2b = (const float*)d_in[31];
  const float* n3g = (const float*)d_in[32]; const float* n3b = (const float*)d_in[33];
  const float* aqw = (const float*)d_in[34]; const float* aqb = (const float*)d_in[35];
  const float* apw = (const float*)d_in[36]; const float* apb = (const float*)d_in[37];
  const float* rqw = (const float*)d_in[38]; const float* rqb = (const float*)d_in[39];
  const float* rpw = (const float*)d_in[40]; const float* rpb = (const float*)d_in[41];
  const float* dqw = (const float*)d_in[42]; const float* dqb = (const float*)d_in[43];
  const float* dpw = (const float*)d_in[44]; const float* dpb = (const float*)d_in[45];
  const float* mw1 = (const float*)d_in[46]; const float* mb1 = (const float*)d_in[47];
  const float* mw2 = (const float*)d_in[48]; const float* mb2 = (const float*)d_in[49];
  const float* op_w1 = (const float*)d_in[50]; const float* op_b1 = (const float*)d_in[51];
  const float* op_g1 = (const float*)d_in[52]; const float* op_be1= (const float*)d_in[53];
  const float* op_w2 = (const float*)d_in[54]; const float* op_b2 = (const float*)d_in[55];

  float* ws = (float*)d_ws;
  const size_t M1 = 1048576;                 // 1M floats
  float* P0   = ws;                          // 8,388,608 floats
  float* P1   = ws + 8388608;                // 8,388,608 floats
  float* FFTO = P0;                          // valid after conv stem is done
  float* X    = P0 + 1*M1;
  float* NX   = P0 + 2*M1;
  float* QKV  = P0 + 3*M1;                   // 3M floats (B,768,2048)
  float* Hbuf = P1;                          // stem output (B,256,64,32)
  float* T0   = P1;                          // reuses Hbuf after X is formed
  float* T1   = P1 + 1*M1;
  float* T2   = P1 + 2*M1;                   // 2M floats (B,512,2048)
  float* OPB  = P1 + 4*M1;                   // (B,128,64,32)
  float* DP   = P1 + 5*M1;

  const int HW = 2048;
  const long NTOT = 2L*256*HW;               // 1,048,576

  dim3 wblk(32, 8);

  // ---- conv stem (ping-pong P1 <-> P0) ----
  transpose_in<<<2048, 256, 0, stream>>>(x, P1);
  conv3d_bn_relu<<<4096, 256, 0, stream>>>(P1, tp_w1, tp_b1, tp_g1, tp_be1, P0,
      2, 2, 4, 128, 256, 16, 4, 128, 256, 1,3,5, 1,1,1, 0,1,2);
  conv3d_bn_relu<<<4096, 256, 0, stream>>>(P0, tp_w2, tp_b2, tp_g2, tp_be2, P1,
      2, 16, 4, 128, 256, 32, 4, 128, 256, 1,3,5, 1,1,1, 0,1,2);
  conv3d_bn_relu<<<4096, 256, 0, stream>>>(P1, tp_w3, tp_b3, tp_g3, tp_be3, P0,
      2, 32, 4, 128, 256, 64, 4, 128, 128, 1,3,5, 1,1,2, 0,1,2);
  conv3d_bn_relu<<<4096, 256, 0, stream>>>(P0, cb_w1, cb_b1, cb_g1, cb_be1, P1,
      2, 64, 4, 128, 128, 64, 1, 128, 128, 4,1,1, 1,1,1, 0,0,0);
  conv3d_bn_relu<<<4096, 256, 0, stream>>>(P1, cb_w2, cb_b2, cb_g2, cb_be2, P0,
      2, 64, 1, 128, 128, 128, 1, 128, 64, 1,1,5, 1,1,2, 0,0,2);
  conv3d_bn_relu<<<4096, 256, 0, stream>>>(P0, cb_w3, cb_b3, cb_g3, cb_be3, Hbuf,
      2, 128, 1, 128, 64, 256, 1, 64, 32, 1,3,3, 1,2,2, 0,1,1);

  // ---- DFT branch + positional embedding ----
  fft_mag<<<512, 256, 0, stream>>>(Hbuf, fft_wr, fft_wd, FFTO);
  add_posemb<<<4096, 256, 0, stream>>>(Hbuf, pos, X, 256, HW, NTOT);

  // ---- transformer blocks ----
  for (int i = 0; i < 3; ++i) {
    const float* qw = aqw + (size_t)i*768*256; const float* qb = aqb + (size_t)i*768;
    const float* pw = apw + (size_t)i*256*256; const float* pb = apb + (size_t)i*256;
    const float* rw = rqw + (size_t)i*768*256; const float* rb = rqb + (size_t)i*768;
    const float* rw2= rpw + (size_t)i*256*256; const float* rb2= rpb + (size_t)i*256;
    const float* dw = dqw + (size_t)i*768*256; const float* db = dqb + (size_t)i*768;
    const float* dw2= dpw + (size_t)i*256*256; const float* db2= dpb + (size_t)i*256;
    const float* w1 = mw1 + (size_t)i*512*256; const float* b1 = mb1 + (size_t)i*512;
    const float* w2 = mw2 + (size_t)i*256*512; const float* b2 = mb2 + (size_t)i*256;

    // MHA sub-block
    gn_ln<<<2, 256, 0, stream>>>(X, n1g + i*256, n1b + i*256, NX, 256, HW);
    gemm_pw<<<dim3(32,12,2), wblk, 0, stream>>>(NX, qw, qb, QKV, 256, 768, HW, 0);
    flash_attn<<<dim3(16,8,2), wblk, 0, stream>>>(QKV, T0, HW);
    gemm_pw<<<dim3(32,4,2), wblk, 0, stream>>>(T0, pw, pb, T1, 256, 256, HW, 0);
    add_ip<<<4096, 256, 0, stream>>>(X, T1, NTOT);

    // axial sub-block (range + doppler on same normed input)
    gn_ln<<<2, 256, 0, stream>>>(X, n2g + i*256, n2b + i*256, NX, 256, HW);
    gemm_pw<<<dim3(32,12,2), wblk, 0, stream>>>(NX, rw, rb, QKV, 256, 768, HW, 0);
    axial_attn<<<512, 256, 0, stream>>>(QKV, T0, HW, 0);
    gemm_pw<<<dim3(32,4,2), wblk, 0, stream>>>(T0, rw2, rb2, T1, 256, 256, HW, 0);
    gemm_pw<<<dim3(32,12,2), wblk, 0, stream>>>(NX, dw, db, QKV, 256, 768, HW, 0);
    axial_attn<<<256, 256, 0, stream>>>(QKV, T0, HW, 1);
    gemm_pw<<<dim3(32,4,2), wblk, 0, stream>>>(T0, dw2, db2, DP, 256, 256, HW, 0);
    add2_ip<<<4096, 256, 0, stream>>>(X, T1, DP, NTOT);

    // MLP sub-block (exact GELU fused into first GEMM epilogue)
    gn_ln<<<2, 256, 0, stream>>>(X, n3g + i*256, n3b + i*256, NX, 256, HW);
    gemm_pw<<<dim3(32,8,2), wblk, 0, stream>>>(NX, w1, b1, T2, 256, 512, HW, 1);
    gemm_pw<<<dim3(32,4,2), wblk, 0, stream>>>(T2, w2, b2, T1, 512, 256, HW, 0);
    add_ip<<<4096, 256, 0, stream>>>(X, T1, NTOT);
  }

  // ---- head ----
  add_ip<<<4096, 256, 0, stream>>>(X, FFTO, NTOT);
  conv3d_bn_relu<<<4096, 256, 0, stream>>>(X, op_w1, op_b1, op_g1, op_be1, OPB,
      2, 256, 1, 64, 32, 128, 1, 64, 32, 1,3,3, 1,1,1, 0,1,1);
  op2_sigmoid<<<16, 256, 0, stream>>>(OPB, op_w2, op_b2, (float*)d_out);
}
